// BiLSTMCRF_19920058319550
// MI455X (gfx1250) — compile-verified, matrix operand order checked by reading, not run on hardware
//
#include <hip/hip_runtime.h>
#include <hip/hip_bf16.h>

typedef __bf16 bf16;
typedef __attribute__((ext_vector_type(16))) __bf16 v16bf;
typedef __attribute__((ext_vector_type(8)))  float  v8f;

#define TSEQ 256
#define BB   128
#define EE   256
#define HH   128
#define HP   136   // scan LDS row stride (bf16): 272B = 68 banks
#define AP   264   // gemm A-stage LDS row stride (bf16): 528B = 132 banks
#define G4   512   // 4*H
#define YC   31
#define YP   32
#define NROW (TSEQ*BB)   // 32768 rows of [T,B] flattened r = t*B + b
#define MTB  8           // M tiles per GEMM block (128 rows)

// ---------------- WMMA helpers (layouts per CDNA5 ISA 7.12.2) ----------------

__device__ inline v8f wmma_bf16(v16bf a, v16bf b, v8f c) {
  return __builtin_amdgcn_wmma_f32_16x16x32_bf16(false, a, false, b, (short)0, c,
                                                 false, false);
}

// Load a 16x32 bf16 fragment from a row-major matrix (works for A tiles and,
// by symmetry, for B tiles sourced from row-major W[N][K] = B^T).
// lane 0-15 : row = lane,     K in {0..7, 16..23}
// lane 16-31: row = lane-16,  K in {8..15, 24..31}
__device__ inline v16bf load_frag16(const bf16* __restrict__ p, int ld, int lane) {
  v16bf r;
  int row = lane & 15;
  int kb  = (lane >> 4) << 3;
  const bf16* rp = p + row * ld;
#pragma unroll
  for (int h = 0; h < 2; ++h) {
#pragma unroll
    for (int w = 0; w < 4; ++w) {
      int k = h * 16 + kb + 2 * w;
      r[h * 8 + 2 * w]     = rp[k];
      r[h * 8 + 2 * w + 1] = rp[k + 1];
    }
  }
  return r;
}

__device__ inline float sigf(float x) { return 1.0f / (1.0f + __expf(-x)); }

// ---------------- weight conversion f32 -> bf16 ----------------

__global__ void k_convert(const float* __restrict__ wih, const float* __restrict__ whh,
                          bf16* __restrict__ wihb, bf16* __restrict__ whhb) {
  int g = blockIdx.x * 256 + threadIdx.x;
  if (g < 3 * 2 * G4 * EE) wihb[g] = (bf16)wih[g];
  if (g < 3 * 2 * G4 * HH) whhb[g] = (bf16)whh[g];
}

// ---------------- embedding gather -> x[T*B, E] bf16 (4 elems/thread) -------

__global__ void k_embed(const int* __restrict__ content, const float* __restrict__ emb,
                        bf16* __restrict__ x) {
  size_t q = (size_t)blockIdx.x * 256 + threadIdx.x;  // over NROW*EE/4
  int k4 = (int)(q & (EE / 4 - 1));   // 64 quads per row
  int r  = (int)(q >> 6);             // r = t*B + b
  int t = r >> 7;
  int b = r & (BB - 1);
  int tok = content[b * TSEQ + t];
  const float4 v = *(const float4*)(emb + (size_t)tok * EE + k4 * 4);
  bf16* xo = x + (size_t)r * EE + k4 * 4;
  xo[0] = (bf16)v.x; xo[1] = (bf16)v.y; xo[2] = (bf16)v.z; xo[3] = (bf16)v.w;
}

// ---------------- fused input GEMM for both directions ----------------
// xg{F,B} = x @ Wih{F,B}^T + (bih + bhh).  M = NROW, N = 512 per dir, K = 256.
// Block (8 waves, 256 thr) owns 8 M-tiles: A (128x256 bf16) is staged once in
// padded LDS and shared by all waves.  Each wave owns one (N-tile, direction)
// pair, so every global B-fragment load feeds 8 independent WMMAs over the 8
// M-tiles: ~55 FLOP per L2 byte -> compute-bound.

__global__ __launch_bounds__(256)
void k_input_gemm(const bf16* __restrict__ xin,
                  const bf16* __restrict__ wF, const bf16* __restrict__ wB,
                  const float* __restrict__ biF, const float* __restrict__ bhF,
                  const float* __restrict__ biB, const float* __restrict__ bhB,
                  float* __restrict__ xgF, float* __restrict__ xgB) {
  __shared__ bf16 ash[MTB * 16 * AP];   // 128 x 256, padded stride
  int lane = threadIdx.x & 31;
  int wave = threadIdx.x >> 5;
  int mtg = blockIdx.x >> 3;   // 256 groups of 8 M-tiles
  int ngo = blockIdx.x & 7;    // 8 N-groups of 4 tiles

  // cooperative A staging: 128 rows x 256 cols, 16B chunks, coalesced
  const bf16* asrc = xin + (size_t)mtg * (MTB * 16) * EE;
  for (int c = threadIdx.x; c < MTB * 16 * EE / 8; c += 256) {
    int row = c >> 5;            // 32 chunks per row
    int c8  = (c & 31) * 8;
    *(uint4*)(ash + row * AP + c8) = *(const uint4*)(asrc + (size_t)row * EE + c8);
  }
  __syncthreads();

  int nt   = ngo * 4 + (wave & 3);
  int dirw = wave >> 2;
  const bf16*  wsrc = dirw ? wB : wF;
  const float* bi   = dirw ? biB : biF;
  const float* bh   = dirw ? bhB : bhF;
  float*       xg   = dirw ? xgB : xgF;

  v8f acc[MTB];
#pragma unroll
  for (int mi = 0; mi < MTB; ++mi) acc[mi] = (v8f){};
  const bf16* b0 = wsrc + (size_t)nt * 16 * EE;
  for (int kk = 0; kk < EE; kk += 32) {
    v16bf b = load_frag16(b0 + kk, EE, lane);
#pragma unroll
    for (int mi = 0; mi < MTB; ++mi) {
      v16bf a = load_frag16(ash + (mi * 16) * AP + kk, AP, lane);
      acc[mi] = wmma_bf16(a, b, acc[mi]);
    }
  }
  int col  = nt * 16 + (lane & 15);
  float bias = bi[col] + bh[col];
#pragma unroll
  for (int mi = 0; mi < MTB; ++mi) {
    int rbase = mtg * (MTB * 16) + mi * 16 + ((lane >> 4) << 3);
#pragma unroll
    for (int v = 0; v < 8; ++v)
      xg[(size_t)(rbase + v) * G4 + col] = acc[mi][v] + bias;
  }
}

// ---------------- persistent LSTM scan (one workgroup per direction) --------
// 32 waves; job grid = 8 (M tiles of B) x 8 (N tiles of H); each wave owns 2
// jobs and all 4 gate tiles of each so the cell update is wave-local.
// h (128x136 bf16, 34KB) AND Whh (512x136 bf16, 136KB) both live in padded
// LDS (170KB total; ISA allows 320KB/WG) so the serial per-step GEMM never
// touches L2.  c stays in registers for all 256 steps.

__global__ __launch_bounds__(1024)
void k_lstm_scan(const float* __restrict__ xgF, const float* __restrict__ xgB,
                 const bf16* __restrict__ whh_l,  // [2][512][128] bf16
                 bf16* __restrict__ xout) {       // [T*B, 256] (concat f|b)
  __shared__ bf16 hsh[HH * HP];
  __shared__ bf16 wsh[G4 * HP];
  int dir = blockIdx.x;
  const float* xg  = dir ? xgB : xgF;
  const bf16*  whh = whh_l + (size_t)dir * G4 * HH;
  int lane = threadIdx.x & 31;
  int wave = threadIdx.x >> 5;

  for (int i = threadIdx.x; i < HH * HP / 8; i += 1024)
    ((uint4*)hsh)[i] = (uint4){0u, 0u, 0u, 0u};
  for (int i = threadIdx.x; i < G4 * HH / 8; i += 1024) {
    int row = i >> 4;
    int c8  = (i & 15) * 8;
    *(uint4*)(wsh + row * HP + c8) = *(const uint4*)(whh + (size_t)row * HH + c8);
  }
  float creg[2][8];
#pragma unroll
  for (int j = 0; j < 2; ++j)
#pragma unroll
    for (int v = 0; v < 8; ++v) creg[j][v] = 0.0f;
  __syncthreads();

  for (int t = 0; t < TSEQ; ++t) {
    int tt = dir ? (TSEQ - 1 - t) : t;
    const float* xgrow = xg + (size_t)tt * BB * G4;
    float hval[2][8];
#pragma unroll
    for (int j = 0; j < 2; ++j) {
      int job = wave * 2 + j;
      int m  = job >> 3;
      int nh = job & 7;
      v8f a0 = {}, a1 = {}, a2 = {}, a3 = {};
#pragma unroll
      for (int kk = 0; kk < HH; kk += 32) {
        v16bf af = load_frag16(hsh + m * 16 * HP + kk, HP, lane);
        v16bf b0 = load_frag16(wsh + (0 * HH + nh * 16) * HP + kk, HP, lane);
        v16bf b1 = load_frag16(wsh + (1 * HH + nh * 16) * HP + kk, HP, lane);
        v16bf b2 = load_frag16(wsh + (2 * HH + nh * 16) * HP + kk, HP, lane);
        v16bf b3 = load_frag16(wsh + (3 * HH + nh * 16) * HP + kk, HP, lane);
        a0 = wmma_bf16(af, b0, a0);
        a1 = wmma_bf16(af, b1, a1);
        a2 = wmma_bf16(af, b2, a2);
        a3 = wmma_bf16(af, b3, a3);
      }
      int col   = nh * 16 + (lane & 15);
      int rbase = m * 16 + ((lane >> 4) << 3);
#pragma unroll
      for (int v = 0; v < 8; ++v) {
        const float* xr = xgrow + (size_t)(rbase + v) * G4;
        float ip = a0[v] + xr[col];
        float fp = a1[v] + xr[HH + col];
        float gp = a2[v] + xr[2 * HH + col];
        float op = a3[v] + xr[3 * HH + col];
        float ii = sigf(ip), ff = sigf(fp), oo = sigf(op);
        float gg = tanhf(gp);
        float c  = ff * creg[j][v] + ii * gg;
        creg[j][v] = c;
        hval[j][v] = oo * tanhf(c);
      }
    }
    __syncthreads();   // all reads of h_prev done before overwrite
#pragma unroll
    for (int j = 0; j < 2; ++j) {
      int job = wave * 2 + j;
      int m = job >> 3, nh = job & 7;
      int col   = nh * 16 + (lane & 15);
      int rbase = m * 16 + ((lane >> 4) << 3);
#pragma unroll
      for (int v = 0; v < 8; ++v) {
        bf16 hb = (bf16)hval[j][v];
        hsh[(rbase + v) * HP + col] = hb;
        xout[(size_t)(tt * BB + rbase + v) * EE + dir * HH + col] = hb;
      }
    }
    __syncthreads();
  }
}

// ---------------- logits + emissions[B][T][32] (Y padded) ----------------

__global__ void k_logits(const bf16* __restrict__ x, const float* __restrict__ W,
                         const float* __restrict__ bias, float* __restrict__ emis) {
  int g = blockIdx.x * 128 + threadIdx.x;  // over NROW*YC
  if (g >= NROW * YC) return;
  int y = g % YC;
  int r = g / YC;
  int t = r >> 7, b = r & (BB - 1);
  float acc = bias[y];
  const bf16*  xr = x + (size_t)r * EE;
  const float* wr = W + (size_t)y * EE;
#pragma unroll 8
  for (int k = 0; k < EE; ++k) acc += (float)xr[k] * wr[k];
  emis[((size_t)b * TSEQ + t) * YP + y] = acc;
}

// ---------------- CRF forward loglik: one block (1 wave) per batch row ------

__global__ __launch_bounds__(32)
void k_crf_fwd(const float* __restrict__ emis, const int* __restrict__ tags,
               const int* __restrict__ mask, const float* __restrict__ trans,
               const float* __restrict__ startv, const float* __restrict__ endv,
               float* __restrict__ partial) {
  __shared__ float alpha[YP];
  int b = blockIdx.x;
  int j = threadIdx.x;
  const float* eb = emis + (size_t)b * TSEQ * YP;
  if (j < YC) alpha[j] = startv[j] + eb[j];
  __syncthreads();
  for (int t = 1; t < TSEQ; ++t) {
    float na = 0.0f;
    int mt = mask[b * TSEQ + t];
    if (j < YC) {
      float mmax = -1e30f;
      for (int i = 0; i < YC; ++i) mmax = fmaxf(mmax, alpha[i] + trans[i * YC + j]);
      float s = 0.0f;
      for (int i = 0; i < YC; ++i) s += __expf(alpha[i] + trans[i * YC + j] - mmax);
      na = mmax + __logf(s) + eb[t * YP + j];
    }
    __syncthreads();
    if (j < YC && mt) alpha[j] = na;
    __syncthreads();
  }
  if (j == 0) {
    float mmax = -1e30f;
    for (int y = 0; y < YC; ++y) mmax = fmaxf(mmax, alpha[y] + endv[y]);
    float s = 0.0f;
    for (int y = 0; y < YC; ++y) s += __expf(alpha[y] + endv[y] - mmax);
    float denom = mmax + __logf(s);
    const int* tb = tags + b * TSEQ;
    float num = startv[tb[0]];
    int cnt = 0;
    for (int t = 0; t < TSEQ; ++t) {
      int mt = mask[b * TSEQ + t];
      float mf = mt ? 1.0f : 0.0f;
      num += eb[t * YP + tb[t]] * mf;
      if (t > 0) num += trans[tb[t - 1] * YC + tb[t]] * mf;
      cnt += mt ? 1 : 0;
    }
    int li = cnt - 1;
    num += endv[tb[li]];
    partial[b] = num - denom;
  }
}

__global__ void k_loss_reduce(const float* __restrict__ partial, float* __restrict__ out) {
  float s = 0.0f;
  for (int b = 0; b < BB; ++b) s += partial[b];
  out[0] = -s;
}

// ---------------- Viterbi: one block (1 wave) per batch row ----------------

__global__ __launch_bounds__(32)
void k_viterbi(const float* __restrict__ emis, const int* __restrict__ mask,
               const float* __restrict__ trans, const float* __restrict__ startv,
               const float* __restrict__ endv, unsigned char* __restrict__ bp,
               float* __restrict__ pred) {
  __shared__ float alpha[YP];
  int b = blockIdx.x;
  int j = threadIdx.x;
  const float* eb = emis + (size_t)b * TSEQ * YP;
  unsigned char* bpb = bp + (size_t)b * TSEQ * YP;
  if (j < YC) alpha[j] = startv[j] + eb[j];
  __syncthreads();
  for (int t = 1; t < TSEQ; ++t) {
    float na = 0.0f;
    int bi = j;
    int mt = mask[b * TSEQ + t];
    if (j < YC) {
      float best = -1e30f;
      int bidx = 0;
      for (int i = 0; i < YC; ++i) {
        float sc = alpha[i] + trans[i * YC + j];
        if (sc > best) { best = sc; bidx = i; }
      }
      na = best + eb[t * YP + j];
      bi = mt ? bidx : j;   // identity when masked out
    }
    __syncthreads();
    if (j < YC) {
      if (mt) alpha[j] = na;
      bpb[t * YP + j] = (unsigned char)bi;
    }
    __syncthreads();
  }
  if (j == 0) {
    float best = -1e30f;
    int tag = 0;
    for (int y = 0; y < YC; ++y) {
      float sc = alpha[y] + endv[y];
      if (sc > best) { best = sc; tag = y; }
    }
    float* pr = pred + (size_t)b * TSEQ;
    pr[TSEQ - 1] = (float)tag;
    for (int t = TSEQ - 2; t >= 0; --t) {
      tag = (int)bpb[(t + 1) * YP + tag];
      pr[t] = (float)tag;
    }
  }
}

// ---------------- host orchestration ----------------

static inline size_t align256(size_t x) { return (x + 255) & ~(size_t)255; }

extern "C" void kernel_launch(void* const* d_in, const int* in_sizes, int n_in,
                              void* d_out, int out_size, void* d_ws, size_t ws_size,
                              hipStream_t stream) {
  const int*   content = (const int*)d_in[0];
  const int*   ner     = (const int*)d_in[1];
  const int*   mask    = (const int*)d_in[2];
  const float* emb     = (const float*)d_in[3];
  const float* Wih     = (const float*)d_in[4];
  const float* Whh     = (const float*)d_in[5];
  const float* bih     = (const float*)d_in[6];
  const float* bhh     = (const float*)d_in[7];
  const float* Wh2     = (const float*)d_in[8];
  const float* bh2     = (const float*)d_in[9];
  const float* trans   = (const float*)d_in[10];
  const float* startv  = (const float*)d_in[11];
  const float* endv    = (const float*)d_in[12];
  float* out = (float*)d_out;

  char* base = (char*)d_ws;
  size_t off = 0;
  auto alloc = [&](size_t bytes) -> void* {
    void* r = base + off;
    off = align256(off + bytes);
    return r;
  };
  bf16*  wihb   = (bf16*)alloc((size_t)3 * 2 * G4 * EE * sizeof(bf16));
  bf16*  whhb   = (bf16*)alloc((size_t)3 * 2 * G4 * HH * sizeof(bf16));
  bf16*  xA     = (bf16*)alloc((size_t)NROW * EE * sizeof(bf16));
  bf16*  xB     = (bf16*)alloc((size_t)NROW * EE * sizeof(bf16));
  float* xgF    = (float*)alloc((size_t)NROW * G4 * sizeof(float));
  float* xgB    = (float*)alloc((size_t)NROW * G4 * sizeof(float));
  float* emis   = (float*)alloc((size_t)BB * TSEQ * YP * sizeof(float));
  float* part   = (float*)alloc((size_t)BB * sizeof(float));
  unsigned char* bpws = (unsigned char*)alloc((size_t)BB * TSEQ * YP);
  (void)ws_size; (void)in_sizes; (void)n_in; (void)out_size;

  k_convert<<<(3 * 2 * G4 * EE + 255) / 256, 256, 0, stream>>>(Wih, Whh, wihb, whhb);
  k_embed<<<(NROW * EE / 4) / 256, 256, 0, stream>>>(content, emb, xA);

  bf16* xin = xA;
  bf16* xout = xB;
  for (int l = 0; l < 3; ++l) {
    const bf16*  wF  = wihb + (size_t)(l * 2 + 0) * G4 * EE;
    const bf16*  wB  = wihb + (size_t)(l * 2 + 1) * G4 * EE;
    const float* biF = bih + (size_t)(l * 2 + 0) * G4;
    const float* bhF = bhh + (size_t)(l * 2 + 0) * G4;
    const float* biB = bih + (size_t)(l * 2 + 1) * G4;
    const float* bhB = bhh + (size_t)(l * 2 + 1) * G4;
    // grid: (NROW/128) M-groups x 8 N-groups, 256 threads
    k_input_gemm<<<(NROW / (MTB * 16)) * 8, 256, 0, stream>>>(
        xin, wF, wB, biF, bhF, biB, bhB, xgF, xgB);
    k_lstm_scan<<<2, 1024, 0, stream>>>(xgF, xgB,
                                        whhb + (size_t)(l * 2) * G4 * HH, xout);
    bf16* tmp = xin; xin = xout; xout = tmp;
  }

  k_logits<<<(NROW * YC + 127) / 128, 128, 0, stream>>>(xin, Wh2, bh2, emis);
  k_crf_fwd<<<BB, 32, 0, stream>>>(emis, ner, mask, trans, startv, endv, part);
  k_loss_reduce<<<1, 1, 0, stream>>>(part, out);
  k_viterbi<<<BB, 32, 0, stream>>>(emis, mask, trans, startv, endv, bpws, out + 1);
}
